// Katies_NeuralSolver_66718021976437
// MI455X (gfx1250) — matrix operand order repaired and checked by reading, compiled, not run on hardware
//
#include <hip/hip_runtime.h>

#define NPTS   81920
#define NB     4
#define DLAT   32
#define HID    128
#define DDYN   16
#define NTILE  (NPTS / 16)           // 5120 point-tiles per batch image
#define WPB    8                     // waves per block
#define WFRAG_HALVES (16384 + 2048)  // W1t + W2t fragment halves (contiguous)

typedef _Float16 v16h __attribute__((ext_vector_type(16)));
typedef _Float16 v8h  __attribute__((ext_vector_type(8)));
typedef float    v8f  __attribute__((ext_vector_type(8)));
typedef unsigned int u32x4 __attribute__((ext_vector_type(4)));
typedef int          i32x4 __attribute__((ext_vector_type(4)));
typedef int          i32x8 __attribute__((ext_vector_type(8)));

static __device__ __forceinline__ v16h cat8(v8h lo, v8h hi) {
  return __builtin_shufflevector(lo, hi, 0,1,2,3,4,5,6,7,8,9,10,11,12,13,14,15);
}

// 8 back-to-back hardware tanh (TRANS ops cover each other's RAW hazard;
// single trailing v_nop guards the last one).
static __device__ __forceinline__ v8f tanh8(v8f x) {
  float r0,r1,r2,r3,r4,r5,r6,r7;
  asm("v_tanh_f32 %0, %8\n\tv_tanh_f32 %1, %9\n\tv_tanh_f32 %2, %10\n\t"
      "v_tanh_f32 %3, %11\n\tv_tanh_f32 %4, %12\n\tv_tanh_f32 %5, %13\n\t"
      "v_tanh_f32 %6, %14\n\tv_tanh_f32 %7, %15\n\tv_nop"
      : "=&v"(r0),"=&v"(r1),"=&v"(r2),"=&v"(r3),
        "=&v"(r4),"=&v"(r5),"=&v"(r6),"=&v"(r7)
      : "v"(x[0]),"v"(x[1]),"v"(x[2]),"v"(x[3]),
        "v"(x[4]),"v"(x[5]),"v"(x[6]),"v"(x[7]));
  v8f r; r[0]=r0; r[1]=r1; r[2]=r2; r[3]=r3; r[4]=r4; r[5]=r5; r[6]=r6; r[7]=r7;
  return r;
}

// B-fragment of flat^T for one source row: per lane 16 contiguous floats at
// d = hf*16 .. hf*16+15  (B layout: lane = column, K split by lane half).
static __device__ __forceinline__ v16h load_b_frag(const float* __restrict__ row, int hf) {
  const float* s = row + hf * 16;
  float4 u0 = *(const float4*)(s);
  float4 u1 = *(const float4*)(s + 4);
  float4 u2 = *(const float4*)(s + 8);
  float4 u3 = *(const float4*)(s + 12);
  v16h a;
  a[0]  = (_Float16)u0.x; a[1]  = (_Float16)u0.y; a[2]  = (_Float16)u0.z; a[3]  = (_Float16)u0.w;
  a[4]  = (_Float16)u1.x; a[5]  = (_Float16)u1.y; a[6]  = (_Float16)u1.z; a[7]  = (_Float16)u1.w;
  a[8]  = (_Float16)u2.x; a[9]  = (_Float16)u2.y; a[10] = (_Float16)u2.z; a[11] = (_Float16)u2.w;
  a[12] = (_Float16)u3.x; a[13] = (_Float16)u3.y; a[14] = (_Float16)u3.z; a[15] = (_Float16)u3.w;
  return a;
}

// hidden permutation sigma: physical WMMA output slot o -> logical hidden unit
// (chosen so each lane's layer-1 D registers ARE its layer-2 B-fragment halves)
static __device__ __host__ inline int sigma_hidden(int o) {
  int kbh  = o >> 5;
  int o_in = o & 31;
  int l_in = (o_in < 16) ? ((o_in >> 3) * 16 + (o_in & 7))
                         : (((o_in - 16) >> 3) * 16 + 8 + ((o_in - 16) & 7));
  return kbh * 32 + l_in;
}

// ---------------------------------------------------------------------------
// Weight pre-swizzle (one-time): W1t = A-fragments of W1^T (hidden columns
// permuted by sigma, flat-K permuted so K_new = c*32 + d), W2t = A-fragments
// of W2^T (logical hidden K).  Emitted CONTIGUOUSLY so one TDM descriptor can
// DMA everything to LDS.
// ---------------------------------------------------------------------------
__global__ void prep_weights(const float* __restrict__ W1,
                             const float* __restrict__ W2,
                             _Float16* __restrict__ W1t,
                             _Float16* __restrict__ W2t) {
  int idx = blockIdx.x * blockDim.x + threadIdx.x;
  if (idx < 16384) {                       // W1t: mt(8) x kf(4) x lane(32) x j(16)
    int mt   = idx >> 11;
    int kf   = (idx >> 9) & 3;             // kf == source c (0=self, 1..3=nbr)
    int lane = (idx >> 4) & 31;
    int j    = idx & 15;
    int hf   = lane >> 4;
    int d    = hf * 16 + j;                // A-layout K
    int Kold = 4 * d + kf;                 // original interleaved flat index
    int col  = sigma_hidden(mt * 16 + (lane & 15));  // permuted hidden column
    W1t[idx] = (_Float16)W1[Kold * HID + col];
  } else if (idx < WFRAG_HALVES) {         // W2t: kb(4) x lane(32) x j(16)
    int t    = idx - 16384;
    int kb   = t >> 9;
    int lane = (t >> 4) & 31;
    int j    = t & 15;
    int hf   = lane >> 4;
    int K    = kb * 32 + hf * 16 + j;
    int o    = lane & 15;                  // dyn output channel
    W2t[t] = (_Float16)W2[K * DDYN + o];
  }
}

// ---------------------------------------------------------------------------
// One Euler step:  z_new[:, :16] = z[:, :16] + tanh(flat @ W1 + b1) @ W2 + b2
// computed transposed:  h^T = W1^T @ flat^T ;  F^T = W2^T @ h^T.
// Weights arrive in LDS via one Tensor-Data-Mover DMA (TENSORcnt-tracked).
// ---------------------------------------------------------------------------
__global__ void __launch_bounds__(256)
solver_step(const float* __restrict__ zsrc, float* __restrict__ zdst,
            const int* __restrict__ nbr,
            const _Float16* __restrict__ Wfrag,   // W1t (16384) ++ W2t (2048)
            const float* __restrict__ b1, const float* __restrict__ b2) {
  __shared__ __align__(16) _Float16 sW[WFRAG_HALVES];  // 36 KB fragment cache

  const int tid  = threadIdx.x;
  const int wave = tid >> 5;
  const int lane = tid & 31;

#if __has_builtin(__builtin_amdgcn_tensor_load_to_lds)
  // ---- TDM: one 1-D descriptor DMAs all 36864 B of fragments into LDS ----
  if (tid == 0) {
    const unsigned nelem = WFRAG_HALVES * 2 / 8;            // 4608 x 8B elements
    unsigned ldsoff = (unsigned)(unsigned long long)
        (__attribute__((address_space(3))) const _Float16*)sW;
    unsigned long long ga = (unsigned long long)Wfrag;
    u32x4 g0;
    g0[0] = 1u;                                             // count=1, user mode
    g0[1] = ldsoff;                                         // lds_addr (bytes)
    g0[2] = (unsigned)ga;                                   // global_addr[95:64]
    g0[3] = (unsigned)((ga >> 32) & 0x01FFFFFFu) | (2u << 30); // addr hi | type=2
    i32x8 g1;
    g1[0] = (int)(3u << 16);                                // data_size = 8B
    g1[1] = (int)((nelem & 0xFFFFu) << 16);                 // tensor_dim0 lo16
    g1[2] = (int)((nelem >> 16) | (1u << 16));              // td0 hi16 | td1=1
    g1[3] = (int)(nelem << 16);                             // tile_dim0
    g1[4] = 0;                                              // tile_dim1/2 unused
    g1[5] = (int)nelem;                                     // tensor_dim0_stride
    g1[6] = 0;
    g1[7] = 0;
    i32x4 gz = {0, 0, 0, 0};
#if defined(__clang_major__) && (__clang_major__ >= 23)
    i32x8 gz8 = {0, 0, 0, 0, 0, 0, 0, 0};
    __builtin_amdgcn_tensor_load_to_lds(g0, g1, gz, gz, gz8, 0);
#else
    __builtin_amdgcn_tensor_load_to_lds(g0, g1, gz, gz, 0);
#endif
#if __has_builtin(__builtin_amdgcn_s_wait_tensorcnt)
    __builtin_amdgcn_s_wait_tensorcnt(0);
#else
    asm volatile("s_wait_tensorcnt 0x0" ::: "memory");
#endif
  }
#else
  // ---- fallback: cooperative copy ----
  {
    const uint4* s1 = (const uint4*)Wfrag;  uint4* d1 = (uint4*)sW;
    for (int i = tid; i < (int)(WFRAG_HALVES * 2 / 16); i += 256) d1[i] = s1[i];
  }
#endif
  __syncthreads();

  const _Float16* sW1 = sW;            // W1^T fragments
  const _Float16* sW2 = sW + 16384;    // W2^T fragments

  const int tile = blockIdx.x * WPB + wave;
  const int b    = tile / NTILE;
  const int p0   = (tile % NTILE) * 16;

  const int m  = lane & 15;   // point within tile (column of all B/D matrices)
  const int hf = lane >> 4;   // lane half
  const int p  = p0 + m;

  const float* zb = zsrc + (size_t)b * NPTS * DLAT;

  // ---- flat^T B-fragments: direct contiguous gather from global ----
  v16h zf[4];
  zf[0] = load_b_frag(zb + (size_t)(unsigned)p * DLAT, hf);
  zf[1] = load_b_frag(zb + (size_t)(unsigned)nbr[p * 3 + 0] * DLAT, hf);
  zf[2] = load_b_frag(zb + (size_t)(unsigned)nbr[p * 3 + 1] * DLAT, hf);
  zf[3] = load_b_frag(zb + (size_t)(unsigned)nbr[p * 3 + 2] * DLAT, hf);

  // ---- layer-2 accumulator: C = z_old[:, hf*8 .. hf*8+7] + b2 (contiguous) --
  v8f acc2;
  {
    const float* zp = zb + (size_t)(unsigned)p * DLAT + hf * 8;
    float4 z0 = *(const float4*)(zp);
    float4 z1 = *(const float4*)(zp + 4);
    float4 q0 = *(const float4*)(b2 + hf * 8);
    float4 q1 = *(const float4*)(b2 + hf * 8 + 4);
    acc2[0] = z0.x + q0.x; acc2[1] = z0.y + q0.y; acc2[2] = z0.z + q0.z; acc2[3] = z0.w + q0.w;
    acc2[4] = z1.x + q1.x; acc2[5] = z1.y + q1.y; acc2[6] = z1.z + q1.z; acc2[7] = z1.w + q1.w;
  }

  // ---- fused layers: per kb, produce hidden tiles 2kb,2kb+1 then consume ----
#pragma unroll
  for (int kb = 0; kb < 4; ++kb) {
    v8h hh[2];
#pragma unroll
    for (int t = 0; t < 2; ++t) {
      const int mt = 2 * kb + t;
      // C init = b1 (hidden-permuted: contiguous 8 floats per lane)
      v8f accH;
      {
        const float* bp = b1 + kb * 32 + hf * 16 + t * 8;
        float4 c0 = *(const float4*)(bp);
        float4 c1 = *(const float4*)(bp + 4);
        accH[0] = c0.x; accH[1] = c0.y; accH[2] = c0.z; accH[3] = c0.w;
        accH[4] = c1.x; accH[5] = c1.y; accH[6] = c1.z; accH[7] = c1.w;
      }
#pragma unroll
      for (int kf = 0; kf < 4; ++kf) {
        const _Float16* aa = &sW1[((mt * 4 + kf) * 32 + lane) * 16];
        v16h af = cat8(*(const v8h*)aa, *(const v8h*)(aa + 8));
        accH = __builtin_amdgcn_wmma_f32_16x16x32_f16(false, af, false, zf[kf],
                                                      (short)0, accH, false, false);
      }
      v8f th = tanh8(accH);
      v8h hv;
      hv[0] = (_Float16)th[0]; hv[1] = (_Float16)th[1];
      hv[2] = (_Float16)th[2]; hv[3] = (_Float16)th[3];
      hv[4] = (_Float16)th[4]; hv[5] = (_Float16)th[5];
      hv[6] = (_Float16)th[6]; hv[7] = (_Float16)th[7];
      hh[t] = hv;
    }
    // layer-2 B-fragment is exactly this lane's two hidden tiles (by sigma)
    v16h b3 = cat8(hh[0], hh[1]);
    const _Float16* aa2 = &sW2[(kb * 32 + lane) * 16];
    v16h a3 = cat8(*(const v8h*)aa2, *(const v8h*)(aa2 + 8));
    acc2 = __builtin_amdgcn_wmma_f32_16x16x32_f16(false, a3, false, b3,
                                                  (short)0, acc2, false, false);
  }

  // ---- store: dyn channels hf*8..hf*8+7 of point p are contiguous ----
  {
    float* op = zdst + ((size_t)b * NPTS + p) * DLAT + hf * 8;
    float4 o0, o1;
    o0.x = acc2[0]; o0.y = acc2[1]; o0.z = acc2[2]; o0.w = acc2[3];
    o1.x = acc2[4]; o1.y = acc2[5]; o1.z = acc2[6]; o1.w = acc2[7];
    *(float4*)(op)     = o0;
    *(float4*)(op + 4) = o1;
  }
  // ---- passthrough channels 16..31 (two lanes per point, 8 floats each) ----
  {
    const size_t cbase = ((size_t)b * NPTS + p) * DLAT + 16 + hf * 8;
    float4 c0 = *(const float4*)(zsrc + cbase);
    float4 c1 = *(const float4*)(zsrc + cbase + 4);
    *(float4*)(zdst + cbase)     = c0;
    *(float4*)(zdst + cbase + 4) = c1;
  }
}

extern "C" void kernel_launch(void* const* d_in, const int* in_sizes, int n_in,
                              void* d_out, int out_size, void* d_ws, size_t ws_size,
                              hipStream_t stream) {
  const float* z0  = (const float*)d_in[0];
  const int*   nbr = (const int*)d_in[1];
  const float* W1  = (const float*)d_in[2];
  const float* b1  = (const float*)d_in[3];
  const float* W2  = (const float*)d_in[4];
  const float* b2  = (const float*)d_in[5];
  float* out = (float*)d_out;

  const size_t zbytes = (size_t)NB * NPTS * DLAT * sizeof(float); // 41,943,040
  char* ws = (char*)d_ws;
  float*    zws = (float*)ws;
  _Float16* W1t = (_Float16*)(ws + zbytes);
  _Float16* W2t = W1t + 16384;          // contiguous with W1t (one TDM blob)

  prep_weights<<<72, 256, 0, stream>>>(W1, W2, W1t, W2t);

  const int blocks = (NB * NTILE) / WPB;   // 2560 blocks x 256 threads
  // ping-pong: in -> ws -> out -> ws -> out
  solver_step<<<blocks, 256, 0, stream>>>(z0,  zws, nbr, W1t, b1, b2);
  solver_step<<<blocks, 256, 0, stream>>>(zws, out, nbr, W1t, b1, b2);
  solver_step<<<blocks, 256, 0, stream>>>(out, zws, nbr, W1t, b1, b2);
  solver_step<<<blocks, 256, 0, stream>>>(zws, out, nbr, W1t, b1, b2);
}
